// OrthoLinear_7773890806217
// MI455X (gfx1250) — matrix operand order repaired and checked by reading, compiled, not run on hardware
//
#include <hip/hip_runtime.h>

// ---------------------------------------------------------------------------
// OrthoLinear: y = x @ (dequant_int4(base,scale) + alpha*CSR_outliers)^T
//   x: [4096 tokens, 4096] fp32    W: [16384, 4096]    out: [4096, 16384] fp32
// Plan (all in d_ws):
//   K0: x fp32 -> f16 Xh (32 MB)
//   K1: dequant int4 -> f16 W (128 MB; fits in 192 MB L2 for GEMM reuse)
//   K2: CSR outlier fold, one thread per nnz, run-head handles duplicates
//   K3: f16 WMMA GEMM, fp32 accumulate, double-buffered LDS fed by
//       GLOBAL_LOAD_ASYNC_TO_LDS_B128 (ASYNCcnt pipeline) when available
// ---------------------------------------------------------------------------

typedef _Float16 v16h __attribute__((ext_vector_type(16)));
typedef _Float16 v8h  __attribute__((ext_vector_type(8)));
typedef float    v8f  __attribute__((ext_vector_type(8)));
typedef int      v4i  __attribute__((ext_vector_type(4)));

typedef __attribute__((address_space(1))) v4i g_v4i;   // global int4
typedef __attribute__((address_space(3))) v4i l_v4i;   // LDS int4

#define IN_F   4096
#define OUT_F  16384
#define KPR    204      // K_PER_ROW (nnz per CSR row)
#define ALPHA  1.0f

#define BM 128
#define BN 128
#define BK 32
#define LDSS 40         // LDS row stride in halves (32 + 8 pad)

#if defined(__has_builtin)
#if __has_builtin(__builtin_amdgcn_global_load_async_to_lds_b128)
#define HAS_ASYNC_LDS 1
#endif
#endif

__device__ __forceinline__ void copy16B_to_lds(const _Float16* g, _Float16* l) {
#ifdef HAS_ASYNC_LDS
    __builtin_amdgcn_global_load_async_to_lds_b128(
        (g_v4i*)g, (l_v4i*)l, 0, 0);
#else
    *(v8h*)l = *(const v8h*)g;          // synchronous fallback
#endif
}

__device__ __forceinline__ void wait_async_prev_stage() {
#ifdef HAS_ASYNC_LDS
#if __has_builtin(__builtin_amdgcn_s_wait_asynccnt)
    __builtin_amdgcn_s_wait_asynccnt((short)4);
#else
    asm volatile("s_wait_asynccnt 0x4" ::: "memory");
#endif
#endif
}

__device__ __forceinline__ void wait_async_zero() {
#ifdef HAS_ASYNC_LDS
#if __has_builtin(__builtin_amdgcn_s_wait_asynccnt)
    __builtin_amdgcn_s_wait_asynccnt((short)0);
#else
    asm volatile("s_wait_asynccnt 0x0" ::: "memory");
#endif
#endif
}

// ---- Kernel 0: x fp32 -> f16 ----------------------------------------------
__global__ __launch_bounds__(256) void xcast_kernel(
    const float* __restrict__ x, _Float16* __restrict__ Xh, int n)
{
    int i = (blockIdx.x * 256 + threadIdx.x) * 8;
    if (i >= n) return;
    float4 f0 = *(const float4*)(x + i);
    float4 f1 = *(const float4*)(x + i + 4);
    v8h h;
    h[0] = (_Float16)f0.x; h[1] = (_Float16)f0.y;
    h[2] = (_Float16)f0.z; h[3] = (_Float16)f0.w;
    h[4] = (_Float16)f1.x; h[5] = (_Float16)f1.y;
    h[6] = (_Float16)f1.z; h[7] = (_Float16)f1.w;
    *(v8h*)(Xh + i) = h;
}

// ---- Kernel 1: unpack int4 (two nibbles per stored int32 byte), scale ------
__global__ __launch_bounds__(256) void dequant_kernel(
    const int* __restrict__ packed,      // [OUT_F, IN_F/2] int32 in [0,256)
    const float* __restrict__ scales,    // [OUT_F]
    _Float16* __restrict__ W)            // [OUT_F, IN_F] f16
{
    unsigned tid = blockIdx.x * 256u + threadIdx.x;
    unsigned j0  = tid * 4u;
    if (j0 >= (unsigned)(OUT_F * (IN_F / 2))) return;
    const uint4 p = *(const uint4*)(packed + j0);
    unsigned row  = j0 >> 11;            // / (IN_F/2)
    unsigned word = j0 & 2047u;
    float s = scales[row];
    v8h o;
    unsigned w;
    w = p.x; o[0] = (_Float16)(((int)( w       & 0xF) - 8) * s);
             o[1] = (_Float16)(((int)((w >> 4) & 0xF) - 8) * s);
    w = p.y; o[2] = (_Float16)(((int)( w       & 0xF) - 8) * s);
             o[3] = (_Float16)(((int)((w >> 4) & 0xF) - 8) * s);
    w = p.z; o[4] = (_Float16)(((int)( w       & 0xF) - 8) * s);
             o[5] = (_Float16)(((int)((w >> 4) & 0xF) - 8) * s);
    w = p.w; o[6] = (_Float16)(((int)( w       & 0xF) - 8) * s);
             o[7] = (_Float16)(((int)((w >> 4) & 0xF) - 8) * s);
    *(v8h*)(W + (size_t)row * IN_F + word * 2u) = o;
}

// ---- Kernel 2: CSR outlier add, one thread per nnz -------------------------
// cols are sorted within each row => duplicates are adjacent runs; only the
// run head writes, summing its run (exactly matches .at[].add accumulation).
__global__ __launch_bounds__(256) void sparse_add_kernel(
    _Float16* __restrict__ W,
    const _Float16* __restrict__ vals,
    const int* __restrict__ cols,
    const int* __restrict__ ptr,
    int nnz)
{
    int j = blockIdx.x * 256 + threadIdx.x;
    if (j >= nnz) return;
    int r = j / KPR;
    int rs = ptr[r], re = ptr[r + 1];
    int c = cols[j];
    if (j > rs && cols[j - 1] == c) return;      // not a run head
    float acc = (float)vals[j];
    for (int jj = j + 1; jj < re && cols[jj] == c; ++jj)
        acc += (float)vals[jj];
    _Float16* wp = W + (size_t)r * IN_F + c;
    *wp = (_Float16)((float)*wp + ALPHA * acc);
}

// ---- Kernel 3: WMMA GEMM  out[m,n] = sum_k Xh[m,k] * W[n,k] ----------------
__global__ __launch_bounds__(256) void gemm_kernel(
    const _Float16* __restrict__ Xh,   // [M, IN_F] f16
    const _Float16* __restrict__ W,    // [OUT_F, IN_F] f16
    float* __restrict__ out)           // [M, OUT_F] fp32
{
    __shared__ _Float16 As[2][BM][LDSS];
    __shared__ _Float16 Bs[2][BN][LDSS];

    const int t    = threadIdx.x;
    const int lane = t & 31;
    const int wid  = t >> 5;
    const int wm   = wid & 1;      // wave M slab (64 rows)
    const int wn   = wid >> 1;     // wave N slab (32 cols)
    const int m16  = lane & 15;
    const int hs   = lane >> 4;    // lane half select

    const int n0 = blockIdx.x * BN;
    const int m0 = blockIdx.y * BM;

    // cooperative loaders: each thread owns 16 contiguous K halves of a row
    const int lrow = t >> 1;
    const int lcb  = (t & 1) * 16;
    const _Float16* agbase = Xh + (size_t)(m0 + lrow) * IN_F + lcb;
    const _Float16* bgbase = W  + (size_t)(n0 + lrow) * IN_F + lcb;

    v8f acc[4][2];
#pragma unroll
    for (int s = 0; s < 4; ++s)
#pragma unroll
        for (int tt = 0; tt < 2; ++tt)
            acc[s][tt] = (v8f)(0.0f);

    // prologue: stage 0 (4 async b128 per thread: 2 for A, 2 for B)
    copy16B_to_lds(agbase,     &As[0][lrow][lcb]);
    copy16B_to_lds(agbase + 8, &As[0][lrow][lcb + 8]);
    copy16B_to_lds(bgbase,     &Bs[0][lrow][lcb]);
    copy16B_to_lds(bgbase + 8, &Bs[0][lrow][lcb + 8]);

    const int nsteps = IN_F / BK;
    for (int it = 0; it < nsteps; ++it) {
        const int buf = it & 1;
        const bool has_next = (it + 1) < nsteps;

        if (has_next) {   // overlap next-stage copy with this stage's WMMAs
            const int kn = (it + 1) * BK;
            copy16B_to_lds(agbase + kn,     &As[buf ^ 1][lrow][lcb]);
            copy16B_to_lds(agbase + kn + 8, &As[buf ^ 1][lrow][lcb + 8]);
            copy16B_to_lds(bgbase + kn,     &Bs[buf ^ 1][lrow][lcb]);
            copy16B_to_lds(bgbase + kn + 8, &Bs[buf ^ 1][lrow][lcb + 8]);
            wait_async_prev_stage();   // asynccnt <= 4: stage `buf` landed
        } else {
            wait_async_zero();
        }
        __syncthreads();               // stage `buf` visible to all waves

        // ---- fragment gathers (wave32 WMMA layouts, ISA 7.12.2) ----
        v16h afrag[4];
#pragma unroll
        for (int s = 0; s < 4; ++s) {
            const _Float16* r = &As[buf][wm * 64 + s * 16 + m16][0];
            v8h lo = *(const v8h*)(r + hs * 8);        // K = hs*8 + 0..7
            v8h hi = *(const v8h*)(r + 16 + hs * 8);   // K = 16 + hs*8 + 0..7
            afrag[s] = __builtin_shufflevector(lo, hi,
                0, 1, 2, 3, 4, 5, 6, 7, 8, 9, 10, 11, 12, 13, 14, 15);
        }
        v16h bfrag[2];
#pragma unroll
        for (int tt = 0; tt < 2; ++tt) {
            const _Float16* r = &Bs[buf][wn * 32 + tt * 16 + m16][0];
            v8h lo = *(const v8h*)(r + hs * 8);        // K = hs*8 + 0..7
            v8h hi = *(const v8h*)(r + 16 + hs * 8);   // K = 16 + hs*8 + 0..7
            bfrag[tt] = __builtin_shufflevector(lo, hi,
                0, 1, 2, 3, 4, 5, 6, 7, 8, 9, 10, 11, 12, 13, 14, 15);
        }

        // ---- 8 WMMAs: 64x32 per wave per K-step ----
#pragma unroll
        for (int s = 0; s < 4; ++s)
#pragma unroll
            for (int tt = 0; tt < 2; ++tt)
                acc[s][tt] = __builtin_amdgcn_wmma_f32_16x16x32_f16(
                    false, afrag[s], false, bfrag[tt],
                    (short)0, acc[s][tt], false, false);

        __syncthreads();   // all waves done reading `buf` before it is reused
    }

    // ---- epilogue: C/D layout — VGPR v: M = v + 8*hs, N = m16 ----
#pragma unroll
    for (int s = 0; s < 4; ++s) {
#pragma unroll
        for (int tt = 0; tt < 2; ++tt) {
            int mrow = m0 + wm * 64 + s * 16 + hs * 8;
            int ncol = n0 + wn * 32 + tt * 16 + m16;
            float* op = out + (size_t)mrow * OUT_F + ncol;
#pragma unroll
            for (int v = 0; v < 8; ++v)
                op[(size_t)v * OUT_F] = acc[s][tt][v];
        }
    }
}

// ---------------------------------------------------------------------------
extern "C" void kernel_launch(void* const* d_in, const int* in_sizes, int n_in,
                              void* d_out, int out_size, void* d_ws, size_t ws_size,
                              hipStream_t stream) {
    const float*    x      = (const float*)d_in[0];
    const int*      packed = (const int*)d_in[1];
    const float*    scales = (const float*)d_in[2];
    const _Float16* vals   = (const _Float16*)d_in[3];
    const int*      cols   = (const int*)d_in[4];
    const int*      ptr    = (const int*)d_in[5];
    float*          out    = (float*)d_out;

    _Float16* W  = (_Float16*)d_ws;                       // 128 MB
    _Float16* Xh = (_Float16*)((char*)d_ws +
                               (size_t)OUT_F * IN_F * sizeof(_Float16));  // +32 MB

    const int xelems = in_sizes[0];          // tokens * IN_F
    const int tokens = xelems / IN_F;        // 4096
    const int nnz    = in_sizes[3];          // OUT_F * KPR

    // K0: x -> f16
    xcast_kernel<<<(xelems / 8 + 255) / 256, 256, 0, stream>>>(x, Xh, xelems);
    // K1: dequant int4 -> f16
    {
        int words = OUT_F * (IN_F / 2);
        dequant_kernel<<<(words / 4) / 256, 256, 0, stream>>>(packed, scales, W);
    }
    // K2: CSR outlier fold (parallel over nnz, run-heads handle duplicates)
    sparse_add_kernel<<<(nnz + 255) / 256, 256, 0, stream>>>(W, vals, cols, ptr, nnz);
    // K3: WMMA GEMM
    {
        dim3 grid(OUT_F / BN, tokens / BM);
        gemm_kernel<<<grid, 256, 0, stream>>>(Xh, W, out);
    }
}
